// MyNet_68659347194297
// MI455X (gfx1250) — matrix operand order repaired and checked by reading, not compiled, not run on hardware
//
#include <hip/hip_runtime.h>

typedef _Float16 v16h __attribute__((ext_vector_type(16)));
typedef _Float16 v8h  __attribute__((ext_vector_type(8)));
typedef _Float16 h8   __attribute__((ext_vector_type(8)));
typedef float    v8f  __attribute__((ext_vector_type(8)));
typedef float    v4f  __attribute__((ext_vector_type(4)));

#define X_GRID 432
#define Y_GRID 496
#define C_OUT  64
#define NPTS   32
#define HPAD   72   // h1 row stride in halves (64+8): 144B, 16B-aligned, conflict-free
#define FPAD   40   // feature row stride in halves (32+8): 80B
#define W1PAD  40   // W1' row stride in halves: 80B (breaks 64B bank cycle)
#define W2PAD  72   // W2' row stride in halves: 144B (breaks 128B bank cycle)

union AB16 { v16h v; h8 h[2]; };

// ---------------------------------------------------------------------------
// Kernel 1: zero the BEV canvas (~219 MB streamed; NT stores to spare L2)
// ---------------------------------------------------------------------------
__global__ void pillar_zero_canvas(v4f* __restrict__ out, long n4) {
    long i = (long)blockIdx.x * blockDim.x + threadIdx.x;
    long stride = (long)gridDim.x * blockDim.x;
    v4f z = {0.f, 0.f, 0.f, 0.f};
    for (; i < n4; i += stride) __builtin_nontemporal_store(z, &out[i]);
}

// ---------------------------------------------------------------------------
// Kernel 2: per-pillar feature build + both MLP layers on WMMA + maxpool +
// scatter. One wave per pillar, 8 waves per block.
//   GEMM1' runs transposed (D = channels x points) so each lane writes 8
//   consecutive H-row halves with a single ds_store_b128; GEMM2 consumes that
//   row-major H directly as A-fragments and pools over rows.
// ---------------------------------------------------------------------------
__global__ __launch_bounds__(256, 2)
void pillar_encoder(const float* __restrict__ pillars,      // (P,32,4)
                    const int*   __restrict__ coors,        // (P,4) b,x,y,_
                    const int*   __restrict__ npoints,      // (P,)
                    const float* __restrict__ W1,           // (64,6)
                    const float* __restrict__ g1, const float* __restrict__ b1,
                    const float* __restrict__ m1, const float* __restrict__ v1,
                    const float* __restrict__ W2,           // (64,64)
                    const float* __restrict__ g2, const float* __restrict__ b2,
                    const float* __restrict__ m2, const float* __restrict__ v2,
                    float* __restrict__ out,                // (B,64,496,432)
                    int P) {
    __shared__ _Float16 s_w1b[C_OUT * W1PAD];     // BN1-scale-folded W1, K-pad 32, [out][k]
    __shared__ _Float16 s_w2t[C_OUT * W2PAD];     // BN2-scale-folded W2, [out][k]
    __shared__ float    s_s1[C_OUT], s_s2[C_OUT]; // BN scales
    __shared__ _Float16 s_b1h[C_OUT];             // BN1-folded bias (f16)
    __shared__ float    s_b2f[C_OUT];             // BN2-folded bias
    __shared__ _Float16 s_feat[8 * NPTS * FPAD];  // per-wave feature tile (f16)
    __shared__ _Float16 s_h[8 * NPTS * HPAD];     // per-wave h1 (f16), [point][ch]

    const int tid = threadIdx.x;

    // ---- BN scale/bias precompute (once), then weight fold into LDS ----
    for (int c = tid; c < C_OUT; c += blockDim.x) {
        float s1 = g1[c] * rsqrtf(v1[c] + 1e-5f);
        float s2 = g2[c] * rsqrtf(v2[c] + 1e-5f);
        s_s1[c] = s1;           s_s2[c] = s2;
        s_b1h[c] = (_Float16)(b1[c] - m1[c] * s1);
        s_b2f[c] = b2[c] - m2[c] * s2;
    }
    __syncthreads();
    for (int i = tid; i < C_OUT * W1PAD; i += blockDim.x) {
        int n = i / W1PAD, k = i % W1PAD;
        s_w1b[i] = (_Float16)((k < 6) ? W1[n * 6 + k] * s_s1[n] : 0.f);
    }
    for (int i = tid; i < C_OUT * W2PAD; i += blockDim.x) {
        int n = i / W2PAD, k = i % W2PAD;
        s_w2t[i] = (_Float16)((k < 64) ? W2[n * 64 + k] * s_s2[n] : 0.f);
    }
    __syncthreads();

    const int wave = tid >> 5;
    const int lane = tid & 31;
    const int p = blockIdx.x * 8 + wave;
    const bool live = (p < P);                    // uniform per wave
    const int halfSel = lane >> 4;                // 0: lanes 0-15, 1: lanes 16-31
    const int l16 = lane & 15;

    int bb = 0, xg = 0, yg = 0;
    _Float16* feat = s_feat + wave * NPTS * FPAD;
    _Float16* hrow = s_h    + wave * NPTS * HPAD;

    if (live) {
        // ---- load point (lane n = point index), fully coalesced 16B/lane ----
        const int n = lane;
        const float4 pt = *(const float4*)(pillars + ((size_t)p * NPTS + n) * 4);
        const int npts = npoints[p];
        bb = coors[p * 4 + 0];
        xg = coors[p * 4 + 1];
        yg = coors[p * 4 + 2];

        // ---- pillar mean over valid points (wave32 xor-shuffle reduction) ----
        const bool valid = (n < npts);
        float sx = valid ? pt.x : 0.f;
        float sy = valid ? pt.y : 0.f;
        float sz = valid ? pt.z : 0.f;
        #pragma unroll
        for (int off = 16; off > 0; off >>= 1) {
            sx += __shfl_xor(sx, off, 32);
            sy += __shfl_xor(sy, off, 32);
            sz += __shfl_xor(sz, off, 32);
        }
        const float inv = 1.0f / (float)npts;

        // ---- 6-channel features (masked), one f16 row (K padded to 32) ----
        float f0 = pt.z;
        float f1 = pt.x - sx * inv;
        float f2 = pt.y - sy * inv;
        float f3 = pt.z - sz * inv;
        float f4 = pt.x - ((float)xg * 0.16f + 0.08f);
        float f5 = pt.y - ((float)yg * 0.16f - 39.60f);
        if (!valid) { f0 = f1 = f2 = f3 = f4 = f5 = 0.f; }

        h8 r0;
        r0[0] = (_Float16)f0; r0[1] = (_Float16)f1;
        r0[2] = (_Float16)f2; r0[3] = (_Float16)f3;
        r0[4] = (_Float16)f4; r0[5] = (_Float16)f5;
        r0[6] = (_Float16)0.f; r0[7] = (_Float16)0.f;
        h8 rz = {};
        *(h8*)&feat[n * FPAD +  0] = r0;
        *(h8*)&feat[n * FPAD +  8] = rz;
        *(h8*)&feat[n * FPAD + 16] = rz;
        *(h8*)&feat[n * FPAD + 24] = rz;
    }
    __syncthreads();   // feature tiles visible

    if (live) {
        // ---- GEMM1' (transposed): D[ch, pt] = W1'(64x32) x F^T(32x32) ----
        // 4 channel tiles (M) x 2 point tiles (N), K=32 in one step.
        // Preload the 2 B fragments (features; lane = point column).
        AB16 bfF[2];
        #pragma unroll
        for (int ptile = 0; ptile < 2; ++ptile) {
            const int n = ptile * 16 + l16;
            bfF[ptile].h[0] = *(const h8*)&feat[n * FPAD + halfSel * 16];
            bfF[ptile].h[1] = *(const h8*)&feat[n * FPAD + halfSel * 16 + 8];
        }
        #pragma unroll
        for (int ot = 0; ot < 4; ++ot) {
            const int o = ot * 16 + l16;          // A row = output channel
            AB16 aw;
            aw.h[0] = *(const h8*)&s_w1b[o * W1PAD + halfSel * 16];
            aw.h[1] = *(const h8*)&s_w1b[o * W1PAD + halfSel * 16 + 8];
            const int rbase = ot * 16 + halfSel * 8;   // channel of D element 0
            const h8 bv = *(const h8*)&s_b1h[rbase];   // broadcast bias (8 ch)
            #pragma unroll
            for (int ptile = 0; ptile < 2; ++ptile) {
                v8h acc = {};
                acc = __builtin_amdgcn_wmma_f16_16x16x32_f16(
                    false, aw.v, false, bfF[ptile].v, (short)0, acc, false, false);
                v8h hz = {};
                v8h hv = __builtin_elementwise_max((v8h)(acc + bv), hz); // BN1+ReLU (packed)
                const int n = ptile * 16 + l16;        // D column = point
                // 8 consecutive channels of H[point] -> single b128 store
                *(h8*)&hrow[n * HPAD + rbase] = hv;
            }
        }
    }
    __syncthreads();   // h1 rows visible

    if (live) {
        // ---- GEMM2: (32x64)·(64x64) via 2x4x2 tiles of wmma_f32_16x16x32_f16
        // Preload all 4 A fragments once (reused across the 4 column tiles)
        AB16 af2[2][2];
        #pragma unroll
        for (int mt = 0; mt < 2; ++mt) {
            const int m = mt * 16 + l16;
            #pragma unroll
            for (int kk = 0; kk < 2; ++kk) {
                const int ka = kk * 32 + halfSel * 16;
                af2[mt][kk].h[0] = *(const h8*)&hrow[m * HPAD + ka];
                af2[mt][kk].h[1] = *(const h8*)&hrow[m * HPAD + ka + 8];
            }
        }
        #pragma unroll 1
        for (int nt = 0; nt < 4; ++nt) {
            const int ncol = nt * 16 + l16;

            // B fragments (32x16 f16): lane = column, contiguous 16 K values
            AB16 bf[2];
            #pragma unroll
            for (int kk = 0; kk < 2; ++kk) {
                const int kb = kk * 32 + halfSel * 16;
                bf[kk].h[0] = *(const h8*)&s_w2t[ncol * W2PAD + kb];
                bf[kk].h[1] = *(const h8*)&s_w2t[ncol * W2PAD + kb + 8];
            }

            float colmax = -3.4e38f;
            #pragma unroll
            for (int mt = 0; mt < 2; ++mt) {
                v8f acc = {};
                #pragma unroll
                for (int kk = 0; kk < 2; ++kk) {
                    acc = __builtin_amdgcn_wmma_f32_16x16x32_f16(
                        false, af2[mt][kk].v, false, bf[kk].v, (short)0, acc, false, false);
                }
                float mrow = acc[0];
                #pragma unroll
                for (int r = 1; r < 8; ++r) mrow = fmaxf(mrow, acc[r]);
                colmax = fmaxf(colmax, mrow);
            }
            // merge row-halves (lane L <-> L^16 share the same column)
            colmax = fmaxf(colmax, __shfl_xor(colmax, 16, 32));
            const float pooled = colmax + s_b2f[ncol];  // BN2 bias commutes with max

            if (lane < 16) {
                // out layout (B, C, Y, X); canvas is streamed, keep it out of L2
                size_t idx = (((size_t)bb * C_OUT + ncol) * Y_GRID + yg) * X_GRID + xg;
                __builtin_nontemporal_store(pooled, &out[idx]);
            }
        }
    }
}

extern "C" void kernel_launch(void* const* d_in, const int* in_sizes, int n_in,
                              void* d_out, int out_size, void* d_ws, size_t ws_size,
                              hipStream_t stream) {
    const float* pillars  = (const float*)d_in[0];
    const int*   coors    = (const int*)  d_in[1];
    const int*   npoints  = (const int*)  d_in[2];
    // d_in[3] = batch_size (scalar) — implied by coors/out layout
    const float* W1 = (const float*)d_in[4];
    const float* g1 = (const float*)d_in[5];
    const float* b1 = (const float*)d_in[6];
    const float* m1 = (const float*)d_in[7];
    const float* v1 = (const float*)d_in[8];
    const float* W2 = (const float*)d_in[9];
    const float* g2 = (const float*)d_in[10];
    const float* b2 = (const float*)d_in[11];
    const float* m2 = (const float*)d_in[12];
    const float* v2 = (const float*)d_in[13];
    float* out = (float*)d_out;

    const int P = in_sizes[0] / (NPTS * 4);   // 64000

    // 1) zero the canvas (reference builds onto a zero canvas)
    long n4 = (long)out_size / 4;
    pillar_zero_canvas<<<2048, 256, 0, stream>>>((v4f*)out, n4);

    // 2) pillar encode + scatter
    int blocks = (P + 7) / 8;
    pillar_encoder<<<blocks, 256, 0, stream>>>(
        pillars, coors, npoints,
        W1, g1, b1, m1, v1,
        W2, g2, b2, m2, v2,
        out, P);
}
